// GraphAttentionLayer_55078660604257
// MI455X (gfx1250) — matrix operand order repaired and checked by reading, compiled, not run on hardware
//
#include <hip/hip_runtime.h>
#include <hip/hip_bf16.h>

typedef __attribute__((ext_vector_type(16))) __bf16 v16bf;
typedef __attribute__((ext_vector_type(8)))  __bf16 v8bf;
typedef __attribute__((ext_vector_type(8)))  float  v8f;

#define B_   128
#define N_   512
#define D_   300
#define A_   32
#define DP_  304   // D padded to multiple of 16 for XbT
#define KP_  320   // D padded to multiple of 32 for projection K-loop

// ---------------------------------------------------------------------------
// Helpers
// ---------------------------------------------------------------------------

__device__ __forceinline__ v8f wmma_bf16(v16bf a, v16bf b, v8f c) {
    return __builtin_amdgcn_wmma_f32_16x16x32_bf16(
        /*neg_a=*/false, a, /*neg_b=*/false, b,
        /*c_mod=*/(short)0, c, /*reuse_a=*/false, /*reuse_b=*/false);
}

// A-layout (16x32 bf16) load from a row-major bf16 buffer:
//   lane l (l<16):  row l,    K = 0..7 (chunk0) and 16..23 (chunk1)
//   lane l (>=16):  row l-16, K = 8..15 and 24..31
// p0 must already point at  rowstart + (lane>>4)*8  (elements).
__device__ __forceinline__ v16bf load_a16(const __bf16* p0) {
    v16bf a;
    reinterpret_cast<v8bf*>(&a)[0] = *reinterpret_cast<const v8bf*>(p0);
    reinterpret_cast<v8bf*>(&a)[1] = *reinterpret_cast<const v8bf*>(p0 + 16);
    return a;
}

// B-layout (32x16 bf16): lane n (n<16) holds column n, K=0..7 and 16..23 (in
// the transposed source row); lane n>=16 holds column n-16, K=8..15 and 24..31.
// p must point at  srcrow(lane&15) + (lane>>4)*8  (elements).
__device__ __forceinline__ v16bf load_b16(const __bf16* p) {
    v16bf b;
    reinterpret_cast<v8bf*>(&b)[0] = *reinterpret_cast<const v8bf*>(p);
    reinterpret_cast<v8bf*>(&b)[1] = *reinterpret_cast<const v8bf*>(p + 16);
    return b;
}

// ---------------------------------------------------------------------------
// Kernel 1: Q = X*Wq, K = X*Wk  (bf16 outputs, WMMA bf16)
// grid: (B*N/16)/4 blocks x 128 threads (4 waves, one 16-row tile per wave)
// ---------------------------------------------------------------------------
__global__ __launch_bounds__(128) void gat_proj_kernel(
    const float* __restrict__ X,  const float* __restrict__ Wq,
    const float* __restrict__ Wk, __bf16* __restrict__ Qb,
    __bf16* __restrict__ Kb)
{
    __shared__ __align__(32) __bf16 WT[2][32 * KP_];  // W^T, zero-padded K

    const int tid = threadIdx.x;
    for (int i = tid; i < 32 * KP_; i += 128) {
        int a = i & 31, k = i >> 5;
        float vq = (k < D_) ? Wq[k * A_ + a] : 0.f;
        float vk = (k < D_) ? Wk[k * A_ + a] : 0.f;
        WT[0][a * KP_ + k] = (__bf16)vq;
        WT[1][a * KP_ + k] = (__bf16)vk;
    }
    __syncthreads();

    const int lane = tid & 31, w = tid >> 5;
    const int bn0  = (blockIdx.x * 4 + w) * 16;     // first row of this tile
    const int r15  = lane & 15, h = lane >> 4;
    const float* xrow = X + (size_t)(bn0 + r15) * D_;

    v8f accQ0 = {}, accQ1 = {}, accK0 = {}, accK1 = {};

    for (int kt = 0; kt < KP_ / 32; ++kt) {
        // A tile: rows of X, f32 -> bf16 in registers, zero-padded tail
        v16bf av;
        __bf16* ap = reinterpret_cast<__bf16*>(&av);
        const int kb0 = kt * 32 + h * 8;
#pragma unroll
        for (int j = 0; j < 8; ++j) {
            int k1 = kb0 + j, k2 = kb0 + 16 + j;
            ap[j]     = (k1 < D_) ? (__bf16)xrow[k1] : (__bf16)0.f;
            ap[j + 8] = (k2 < D_) ? (__bf16)xrow[k2] : (__bf16)0.f;
        }
        const int boff = kt * 32 + h * 8;
        v16bf bq0 = load_b16(&WT[0][(0  + r15) * KP_ + boff]);
        v16bf bq1 = load_b16(&WT[0][(16 + r15) * KP_ + boff]);
        v16bf bk0 = load_b16(&WT[1][(0  + r15) * KP_ + boff]);
        v16bf bk1 = load_b16(&WT[1][(16 + r15) * KP_ + boff]);
        accQ0 = wmma_bf16(av, bq0, accQ0);
        accQ1 = wmma_bf16(av, bq1, accQ1);
        accK0 = wmma_bf16(av, bk0, accK0);
        accK1 = wmma_bf16(av, bk1, accK1);
    }

    // C layout: VGPR r, lane t -> row r + 8*(t>>4), col t&15
#pragma unroll
    for (int r = 0; r < 8; ++r) {
        size_t row = (size_t)(bn0 + r + 8 * h);
        Qb[row * A_ + (0  + r15)] = (__bf16)accQ0[r];
        Qb[row * A_ + (16 + r15)] = (__bf16)accQ1[r];
        Kb[row * A_ + (0  + r15)] = (__bf16)accK0[r];
        Kb[row * A_ + (16 + r15)] = (__bf16)accK1[r];
    }
}

// ---------------------------------------------------------------------------
// Kernel 2: XbT[b][d][n] = bf16(X[b][n][d]), d zero-padded to DP_
// grid: B * (N/32) * 10 blocks x 256 threads, 32x32 LDS tile transpose
// ---------------------------------------------------------------------------
__global__ __launch_bounds__(256) void gat_xpose_kernel(
    const float* __restrict__ X, __bf16* __restrict__ XbT)
{
    __shared__ float T[32][33];
    int blk = blockIdx.x;
    int dt  = blk % 10;  int tmp = blk / 10;
    int nt  = tmp & 15;  int b   = tmp >> 4;
    int n0 = nt * 32, d0 = dt * 32;
    int j = threadIdx.x & 31, i0 = threadIdx.x >> 5;

    for (int i = i0; i < 32; i += 8) {
        int d = d0 + j;
        T[i][j] = (d < D_) ? X[((size_t)b * N_ + n0 + i) * D_ + d] : 0.f;
    }
    __syncthreads();
    for (int dd = i0; dd < 32; dd += 8) {
        int d = d0 + dd;
        if (d < DP_)
            XbT[((size_t)b * DP_ + d) * N_ + n0 + j] = (__bf16)T[j][dd];
    }
}

// ---------------------------------------------------------------------------
// Kernel 3: attention. One block (4 waves) per (b, 16-row n-tile).
//   Phase 1: S = (Q K^T) * Adj   -> LDS f32 [16][512]
//   Phase 2: row softmax; store unnormalized exp as bf16 P + 1/rowsum
//   Phase 3: O = P @ X  via WMMA over XbT; scale by 1/rowsum on store
// ---------------------------------------------------------------------------
__global__ __launch_bounds__(128) void gat_attn_kernel(
    const float* __restrict__ Adj, const __bf16* __restrict__ Qb,
    const __bf16* __restrict__ Kb, const __bf16* __restrict__ XbT,
    float* __restrict__ Out)
{
    __shared__ __align__(32) float  S[16 * N_];    // 32 KB scores
    __shared__ __align__(32) __bf16 P[16 * N_];    // 16 KB probs
    __shared__ float rinv[16];

    const int tid = threadIdx.x, lane = tid & 31, w = tid >> 5;
    const int b  = blockIdx.x >> 5;       // N_/16 = 32 tiles per batch
    const int nt = blockIdx.x & 31;
    const int n0 = nt * 16;
    const int r15 = lane & 15, h = lane >> 4;

    // ---- Phase 1: scores (one bf16 WMMA per 16x16 tile; K-dim == A_ == 32)
    const __bf16* qrow = Qb + (size_t)(b * N_ + n0 + r15) * A_ + h * 8;
    v16bf qa = load_a16(qrow);

    for (int i = 0; i < 8; ++i) {
        int m0 = (w * 8 + i) * 16;
        v16bf kv = load_b16(Kb + (size_t)(b * N_ + m0 + r15) * A_ + h * 8);
        v8f c = {};
        c = wmma_bf16(qa, kv, c);
        int col = m0 + r15;
        const float* arow = Adj + ((size_t)b * N_ + n0 + 8 * h) * N_ + col;
#pragma unroll
        for (int r = 0; r < 8; ++r)
            S[(r + 8 * h) * N_ + col] = c[r] * arow[(size_t)r * N_];
    }
    __syncthreads();

    // ---- Phase 2: softmax over 512 cols; 8 lanes per row
    {
        int row = tid >> 3, sub = tid & 7;
        float mx = -1e30f;
        for (int cidx = sub; cidx < N_; cidx += 8)
            mx = fmaxf(mx, S[row * N_ + cidx]);
        mx = fmaxf(mx, __shfl_xor(mx, 1, 8));
        mx = fmaxf(mx, __shfl_xor(mx, 2, 8));
        mx = fmaxf(mx, __shfl_xor(mx, 4, 8));
        float sum = 0.f;
        for (int cidx = sub; cidx < N_; cidx += 8) {
            float e = __expf(S[row * N_ + cidx] - mx);
            sum += e;
            P[row * N_ + cidx] = (__bf16)e;
        }
        sum += __shfl_xor(sum, 1, 8);
        sum += __shfl_xor(sum, 2, 8);
        sum += __shfl_xor(sum, 4, 8);
        if (sub == 0) rinv[row] = 1.0f / sum;
    }
    __syncthreads();

    // ---- Phase 3: O = P @ X   (19 d-chunks of 16, K = 512 in 16 steps of 32)
    const __bf16* xb_base = XbT + (size_t)b * DP_ * N_;
    for (int dt = w; dt < DP_ / 16; dt += 4) {
        int d0 = dt * 16;
        v8f acc = {};
        const __bf16* xb = xb_base + (size_t)(d0 + r15) * N_ + h * 8;
        const __bf16* pl = P + r15 * N_ + h * 8;
        for (int mt = 0; mt < N_ / 32; ++mt) {
            v16bf pa = load_a16(pl + mt * 32);
            v16bf xv = load_b16(xb + mt * 32);
            acc = wmma_bf16(pa, xv, acc);
        }
        int col = d0 + r15;
        if (col < D_) {
#pragma unroll
            for (int r = 0; r < 8; ++r) {
                int rrow = r + 8 * h;
                Out[((size_t)(b * N_ + n0 + rrow)) * D_ + col] = acc[r] * rinv[rrow];
            }
        }
    }
}

// ---------------------------------------------------------------------------
// Launcher
// ---------------------------------------------------------------------------
extern "C" void kernel_launch(void* const* d_in, const int* in_sizes, int n_in,
                              void* d_out, int out_size, void* d_ws, size_t ws_size,
                              hipStream_t stream) {
    (void)in_sizes; (void)n_in; (void)out_size; (void)ws_size;
    const float* X   = (const float*)d_in[0];
    const float* Adj = (const float*)d_in[1];
    const float* Wq  = (const float*)d_in[2];
    const float* Wk  = (const float*)d_in[3];
    float* Out = (float*)d_out;

    __bf16* Qb  = (__bf16*)d_ws;                       //  4 MB
    __bf16* Kb  = Qb + (size_t)B_ * N_ * A_;           //  4 MB
    __bf16* XbT = Kb + (size_t)B_ * N_ * A_;           // ~38 MB

    gat_proj_kernel<<<(B_ * N_ / 16) / 4, 128, 0, stream>>>(X, Wq, Wk, Qb, Kb);
    gat_xpose_kernel<<<B_ * (N_ / 32) * 10, 256, 0, stream>>>(X, XbT);
    gat_attn_kernel<<<B_ * (N_ / 16), 128, 0, stream>>>(Adj, Qb, Kb, XbT, Out);
}
